// CrossAttention1_43396349559276
// MI455X (gfx1250) — matrix operand order repaired and checked by reading, not compiled, not run on hardware
//
#include <hip/hip_runtime.h>
#include <hip/hip_bf16.h>

// CDNA5 / gfx1250: wave32, v_wmma_f32_16x16x32_bf16, async global->LDS
// double-buffered staging (ASYNCcnt), in-register head-dot reduction.
typedef __bf16 bf16;
typedef __bf16 v16bf __attribute__((ext_vector_type(16)));
typedef float  v8f   __attribute__((ext_vector_type(8)));

#define TM   32      // rows per workgroup in kernel A
#define XLD  1032    // 1024 + 8 bf16 pad
#define WLD  72      // 64 + 8 bf16 pad (row stride 144B, 16B-aligned)

static __device__ __forceinline__ bf16 f2bf(float f) {
  unsigned u = __builtin_bit_cast(unsigned, f);
  unsigned short h = (unsigned short)((u + 0x7FFFu + ((u >> 16) & 1u)) >> 16);
  return __builtin_bit_cast(bf16, h);
}
static __device__ __forceinline__ unsigned pack2(float lo, float hi) {
  unsigned a = __builtin_bit_cast(unsigned, lo);
  unsigned b = __builtin_bit_cast(unsigned, hi);
  a = (a + 0x7FFFu + ((a >> 16) & 1u)) >> 16;
  b = (b + 0x7FFFu + ((b >> 16) & 1u)) & 0xFFFF0000u;
  return a | b;
}

// 16B async copy global->LDS (GVS). Low 32 bits of a generic shared pointer
// are the LDS byte offset.
static __device__ __forceinline__ void
async_b128(void* lds_dst, const void* gbase, unsigned byte_off) {
  unsigned loff = (unsigned)(unsigned long long)lds_dst;
  asm volatile("global_load_async_to_lds_b128 %0, %1, %2 offset:0"
               :: "v"(loff), "v"(byte_off), "s"(gbase) : "memory");
}

// ---------------------------------------------------------------------------
// Prep: W[k][n] fp32 -> WT[n][k] bf16, one-time.
// ---------------------------------------------------------------------------
__global__ __launch_bounds__(256)
void transpose_bf16_kernel(const float* __restrict__ W, bf16* __restrict__ WT,
                           int N)
{
  __shared__ float t[64][68];
  const int tid = threadIdx.x;
  const int n0  = blockIdx.x * 64;
  const int k0  = blockIdx.y * 64;
#pragma unroll
  for (int j = 0; j < 4; ++j) {
    int c4 = tid + j * 256;
    int rl = c4 >> 4, col4 = (c4 & 15) * 4;
    float4 v = *(const float4*)(W + (size_t)(k0 + rl) * N + n0 + col4);
    t[rl][col4 + 0] = v.x; t[rl][col4 + 1] = v.y;
    t[rl][col4 + 2] = v.z; t[rl][col4 + 3] = v.w;
  }
  __syncthreads();
#pragma unroll
  for (int j = 0; j < 2; ++j) {
    int c = tid + j * 256;
    int nl = c >> 3, k8 = (c & 7) * 8;
    uint4 u;
    u.x = pack2(t[k8 + 0][nl], t[k8 + 1][nl]);
    u.y = pack2(t[k8 + 2][nl], t[k8 + 3][nl]);
    u.z = pack2(t[k8 + 4][nl], t[k8 + 5][nl]);
    u.w = pack2(t[k8 + 6][nl], t[k8 + 7][nl]);
    *reinterpret_cast<uint4*>(WT + (size_t)(n0 + nl) * 1024 + k0 + k8) = u;
  }
}

// ---------------------------------------------------------------------------
// Async stage of one 64(N)x64(K) bf16 slab of WT into an LDS buffer.
// 2 async b128 per thread (512 chunks / 256 threads).
// ---------------------------------------------------------------------------
static __device__ __forceinline__ void
stage_w(bf16 (*Wb)[WLD], const bf16* __restrict__ WT, int wrow0, int ko, int tid) {
#pragma unroll
  for (int j = 0; j < 2; ++j) {
    int c  = tid + j * 256;
    int nl = c >> 3, k8 = (c & 7) * 8;
    async_b128(&Wb[nl][k8], WT,
               (unsigned)(((wrow0 + nl) * 1024 + ko * 64 + k8) * 2));
  }
}

// ---------------------------------------------------------------------------
// Per-wave 16x16 tile of A(32x1024 bf16 LDS) @ WT[wrow0..+64][:]^T, K=1024.
// Double-buffered async weight staging; two independent WMMA accumulation
// chains (even/odd 32-K halves) to avoid D->C hazard bubbles.
// ---------------------------------------------------------------------------
static __device__ __forceinline__ v8f
gemm_tile64(const bf16 (*A)[XLD], bf16 W[2][64][WLD],
            const bf16* __restrict__ WT, int wrow0,
            int tid, int mrow, int ncol)
{
  const int lane = tid & 31;
  const int mloc = lane & 15;
  const int hh   = lane >> 4;
  v8f acc0, acc1;
#pragma unroll
  for (int i = 0; i < 8; ++i) { acc0[i] = 0.0f; acc1[i] = 0.0f; }

  stage_w(W[0], WT, wrow0, 0, tid);
  for (int ko = 0; ko < 16; ++ko) {
    const int cur = ko & 1;
    if (ko < 15) {
      stage_w(W[cur ^ 1], WT, wrow0, ko + 1, tid);
      asm volatile("s_wait_asynccnt 2" ::: "memory");   // current slab landed
    } else {
      asm volatile("s_wait_asynccnt 0" ::: "memory");
    }
    __syncthreads();

    const int kb = ko * 64;
    v16bf a0, a1, b0, b1;
#pragma unroll
    for (int e = 0; e < 8; ++e) {            // A 16x32 layout (ISA 7.12.2)
      a0[e]     = A[mrow + mloc][kb + hh * 8 + e];
      a0[e + 8] = A[mrow + mloc][kb + 16 + hh * 8 + e];
      a1[e]     = A[mrow + mloc][kb + 32 + hh * 8 + e];
      a1[e + 8] = A[mrow + mloc][kb + 48 + hh * 8 + e];
    }
#pragma unroll
    for (int e = 0; e < 16; ++e) {           // B 32x16: same K order as A (ISA 7.12.2)
      b0[e] = W[cur][ncol + mloc][hh * 8 + (e & 8) + e];
      b1[e] = W[cur][ncol + mloc][32 + hh * 8 + (e & 8) + e];
    }
    acc0 = __builtin_amdgcn_wmma_f32_16x16x32_bf16(false, a0, false, b0,
                                                   (short)0, acc0, false, false);
    acc1 = __builtin_amdgcn_wmma_f32_16x16x32_bf16(false, a1, false, b1,
                                                   (short)0, acc1, false, false);
    __syncthreads();                         // reads done before buffer reuse
  }
#pragma unroll
  for (int i = 0; i < 8; ++i) acc0[i] += acc1[i];
  return acc0;
}

// ---------------------------------------------------------------------------
// Kernel A: Q,K per head (WMMA) -> in-register dot -> softmax over heads ->
// V per head scaled by attn -> workspace P (bf16).
// ---------------------------------------------------------------------------
__global__ __launch_bounds__(256, 1)
void qkv_attn_kernel(const float* __restrict__ X, const float* __restrict__ Y,
                     const bf16* __restrict__ WqT, const bf16* __restrict__ WkvT,
                     bf16* __restrict__ P)
{
  __shared__ bf16  x_lds[TM][XLD];
  __shared__ bf16  y_lds[TM][XLD];
  __shared__ bf16  w_lds[2][64][WLD];
  __shared__ float sc_lds[TM][16];

  const int tid  = threadIdx.x;
  const int lane = tid & 31;
  const int mloc = lane & 15;
  const int hh   = lane >> 4;
  const int wave = tid >> 5;
  const int r0   = blockIdx.x * TM;
  const int mrow = (wave >> 2) * 16;
  const int ncol = (wave & 3) * 16;

  for (int i = tid; i < TM * 16; i += 256) ((float*)sc_lds)[i] = 0.0f;

  // Stage x,y fp32 -> bf16 LDS (float4 loads, packed dword-pair stores).
#pragma unroll 4
  for (int e = 0; e < 32; ++e) {
    int i4  = tid + e * 256;
    int row = i4 >> 8;
    int c4  = (i4 & 255) << 2;
    float4 vx = ((const float4*)(X + (size_t)(r0 + row) * 1024))[i4 & 255];
    uint2 px; px.x = pack2(vx.x, vx.y); px.y = pack2(vx.z, vx.w);
    *reinterpret_cast<uint2*>(&x_lds[row][c4]) = px;
    float4 vy = ((const float4*)(Y + (size_t)(r0 + row) * 1024))[i4 & 255];
    uint2 py; py.x = pack2(vy.x, vy.y); py.y = pack2(vy.z, vy.w);
    *reinterpret_cast<uint2*>(&y_lds[row][c4]) = py;
  }

  // Phase 1: per-head dots, reduced in-register across the 16 N-lanes.
  for (int h = 0; h < 16; ++h) {
    v8f q = gemm_tile64(x_lds, w_lds, WqT, h * 64, tid, mrow, ncol);
    v8f k = gemm_tile64(y_lds, w_lds, WkvT, h * 64, tid, mrow, ncol);
#pragma unroll
    for (int i = 0; i < 8; ++i) {
      float s = q[i] * k[i];
      s += __shfl_xor(s, 1, 32);
      s += __shfl_xor(s, 2, 32);
      s += __shfl_xor(s, 4, 32);
      s += __shfl_xor(s, 8, 32);
      if (mloc == 0) atomicAdd(&sc_lds[mrow + 8 * hh + i][h], s);
    }
  }
  __syncthreads();

  // Phase 2: softmax over the 16 heads (double scale (1/8)^2 on the dots).
  if (tid < TM) {
    float d[16]; float mx = -3.0e38f;
#pragma unroll
    for (int h = 0; h < 16; ++h) {
      d[h] = sc_lds[tid][h] * 0.015625f;
      mx = fmaxf(mx, d[h]);
    }
    float sum = 0.0f;
#pragma unroll
    for (int h = 0; h < 16; ++h) { d[h] = __expf(d[h] - mx); sum += d[h]; }
    float inv = 1.0f / sum;
#pragma unroll
    for (int h = 0; h < 16; ++h) sc_lds[tid][h] = d[h] * inv;
  }
  __syncthreads();

  // Phase 3: V scaled by attn -> P (bf16).
  for (int h = 0; h < 16; ++h) {
    v8f v = gemm_tile64(y_lds, w_lds, WkvT, 1024 + h * 64, tid, mrow, ncol);
#pragma unroll
    for (int i = 0; i < 8; ++i) {
      int row = mrow + 8 * hh + i;
      float val = v[i] * sc_lds[row][h];
      P[(size_t)(r0 + row) * 1024 + h * 64 + ncol + mloc] = f2bf(val);
    }
  }
}

// ---------------------------------------------------------------------------
// Kernel C: Out = P(bf16) @ Wproj + bproj. 64x64 tile per WG, async
// double-buffered staging, 4 independent WMMA chains per wave.
// ---------------------------------------------------------------------------
__global__ __launch_bounds__(256, 2)
void proj_kernel(const bf16* __restrict__ P, const bf16* __restrict__ WpT,
                 const float* __restrict__ bp, float* __restrict__ Out)
{
  __shared__ bf16 p_lds[2][64][WLD];
  __shared__ bf16 w_lds[2][64][WLD];

  const int tid  = threadIdx.x;
  const int lane = tid & 31;
  const int wave = tid >> 5;
  const int r0   = blockIdx.y * 64;
  const int c0   = blockIdx.x * 64;
  const int mloc = lane & 15;
  const int hh   = lane >> 4;

  auto stage = [&](int ko, int buf) {        // 4 async b128 per thread
#pragma unroll
    for (int j = 0; j < 2; ++j) {
      int c  = tid + j * 256;
      int rl = c >> 3, k8 = (c & 7) * 8;
      async_b128(&p_lds[buf][rl][k8], P,
                 (unsigned)(((r0 + rl) * 1024 + ko * 64 + k8) * 2));
      async_b128(&w_lds[buf][rl][k8], WpT,
                 (unsigned)(((c0 + rl) * 1024 + ko * 64 + k8) * 2));
    }
  };

  v8f acc[2][2];                             // [tile][K-half chain]
#pragma unroll
  for (int t = 0; t < 2; ++t)
#pragma unroll
    for (int p = 0; p < 2; ++p)
#pragma unroll
      for (int i = 0; i < 8; ++i) acc[t][p][i] = 0.0f;

  stage(0, 0);
  for (int ko = 0; ko < 16; ++ko) {
    const int cur = ko & 1;
    if (ko < 15) {
      stage(ko + 1, cur ^ 1);
      asm volatile("s_wait_asynccnt 4" ::: "memory");
    } else {
      asm volatile("s_wait_asynccnt 0" ::: "memory");
    }
    __syncthreads();

#pragma unroll
    for (int t = 0; t < 2; ++t) {
      int tile = wave + t * 8;
      int mrow = (tile >> 2) * 16;
      int ncol = (tile & 3) * 16;
      v16bf a0, a1, b0, b1;
#pragma unroll
      for (int e = 0; e < 8; ++e) {
        a0[e]     = p_lds[cur][mrow + mloc][hh * 8 + e];
        a0[e + 8] = p_lds[cur][mrow + mloc][16 + hh * 8 + e];
        a1[e]     = p_lds[cur][mrow + mloc][32 + hh * 8 + e];
        a1[e + 8] = p_lds[cur][mrow + mloc][48 + hh * 8 + e];
      }
#pragma unroll
      for (int e = 0; e < 16; ++e) {
        b0[e] = w_lds[cur][ncol + mloc][hh * 8 + (e & 8) + e];
        b1[e] = w_lds[cur][ncol + mloc][32 + hh * 8 + (e & 8) + e];
      }
      acc[t][0] = __builtin_amdgcn_wmma_f32_16x16x32_bf16(false, a0, false, b0,
                                                          (short)0, acc[t][0], false, false);
      acc[t][1] = __builtin_amdgcn_wmma_f32_16x16x32_bf16(false, a1, false, b1,
                                                          (short)0, acc[t][1], false, false);
    }
    __syncthreads();
  }

#pragma unroll
  for (int t = 0; t < 2; ++t) {
    int tile = wave + t * 8;
    int mrow = (tile >> 2) * 16;
    int ncol = (tile & 3) * 16;
    float bpv = bp[c0 + ncol + mloc];
#pragma unroll
    for (int i = 0; i < 8; ++i) {
      int row = mrow + 8 * hh + i;
      Out[(size_t)(r0 + row) * 1024 + c0 + ncol + mloc] =
          acc[t][0][i] + acc[t][1][i] + bpv;
    }
  }
}

extern "C" void kernel_launch(void* const* d_in, const int* in_sizes, int n_in,
                              void* d_out, int out_size, void* d_ws, size_t ws_size,
                              hipStream_t stream) {
  (void)in_sizes; (void)n_in; (void)out_size; (void)ws_size;
  const float* X   = (const float*)d_in[0];
  const float* Y   = (const float*)d_in[1];
  // d_in[2] = mask (unused by the module)
  const float* Wq  = (const float*)d_in[3];
  const float* Wkv = (const float*)d_in[4];
  const float* Wp  = (const float*)d_in[5];
  const float* bp  = (const float*)d_in[6];
  float* Out = (float*)d_out;

  // Workspace layout: WqT 2MB | WkvT 4MB | WpT 2MB | P 128MB
  char* ws = (char*)d_ws;
  bf16* WqT  = (bf16*)(ws);
  bf16* WkvT = (bf16*)(ws + (2ull << 20));
  bf16* WpT  = (bf16*)(ws + (6ull << 20));
  bf16* P    = (bf16*)(ws + (8ull << 20));

  transpose_bf16_kernel<<<dim3(16, 16), 256, 0, stream>>>(Wq,  WqT,  1024);
  transpose_bf16_kernel<<<dim3(32, 16), 256, 0, stream>>>(Wkv, WkvT, 2048);
  transpose_bf16_kernel<<<dim3(16, 16), 256, 0, stream>>>(Wp,  WpT,  1024);
  qkv_attn_kernel<<<dim3(65536 / TM), 256, 0, stream>>>(X, Y, WqT, WkvT, P);
  proj_kernel<<<dim3(16, 1024), 256, 0, stream>>>(P, WpT, bp, Out);
}
